// RelationUnit_10943576670544
// MI455X (gfx1250) — compile-verified, matrix operand order checked by reading, not run on hardware
//
#include <hip/hip_runtime.h>
#include <hip/hip_bf16.h>

// ---------------------------------------------------------------------------
// Types for CDNA5 WMMA
// ---------------------------------------------------------------------------
typedef __bf16 bf16;
typedef __attribute__((ext_vector_type(16))) __bf16 v16bf;
typedef __attribute__((ext_vector_type(8)))  float  v8f;

union AFrag {
    v16bf v;
    uint4 u[2];
};
union H4 {
    bf16 h[4];
    unsigned long long q;
};

// ---------------------------------------------------------------------------
// Problem constants
// ---------------------------------------------------------------------------
constexpr int NR    = 4096;   // rois
constexpr int DORI  = 2048;
constexpr int DAPP  = 512;
constexpr int NA    = 9;      // anchors
constexpr int NCLS  = 21;
constexpr int SCOLS = 32;     // small-gemm cols: 4 alpha | 4 offset | 21 cls | 1 alpha_cls | 2 pad
constexpr int SPAD  = 128;    // small panel padded to one 128-wide N tile
constexpr int QCOLS = 160;    // packed q/k cols: 128 (k*4+c) reg | 32 cls

// Workspace layout (bytes)
constexpr size_t WS_OFF    = 0;                                           // bf16 off[10][4096][512]
constexpr size_t WS_OFF_B  = (size_t)(NA + 1) * NR * DAPP * 2;
constexpr size_t WS_S      = WS_OFF + WS_OFF_B;                           // f32 S[9][4096][32]
constexpr size_t WS_S_B    = (size_t)NA * NR * SCOLS * 4;
constexpr size_t WS_Q      = WS_S + WS_S_B;                               // f32 Q[9][4096][160]
constexpr size_t WS_Q_B    = (size_t)NA * NR * QCOLS * 4;
constexpr size_t WS_K      = WS_Q + WS_Q_B;                               // f32 KY[4096][160]
constexpr size_t WS_K_B    = (size_t)NR * QCOLS * 4;
constexpr size_t WS_BATT   = WS_K + WS_K_B;                               // bf16 [2048][512]
constexpr size_t WS_BATT_B = (size_t)DORI * DAPP * 2;
constexpr size_t WS_WSMP   = WS_BATT + WS_BATT_B;                         // bf16 [9][2048][128]
constexpr size_t WS_WSMP_B = (size_t)NA * DORI * SPAD * 2;
constexpr size_t WS_WQ     = WS_WSMP + WS_WSMP_B;                         // bf16 [9][512][160]
constexpr size_t WS_WQ_B   = (size_t)NA * DAPP * QCOLS * 2;
constexpr size_t WS_WK     = WS_WQ + WS_WQ_B;                             // bf16 [512][160]

// Output layout (float element offsets, concatenated in tuple order)
constexpr size_t O_OUT    = 0;                            // (4096,4)
constexpr size_t O_OUTB   = O_OUT    + (size_t)NR * 4;    // (4096,4)
constexpr size_t O_ASM    = O_OUTB   + (size_t)NR * 4;    // (9,4096,4)
constexpr size_t O_BSM    = O_ASM    + (size_t)NA * NR * 4;
constexpr size_t O_D1     = O_BSM    + (size_t)NA * NR * 4;
constexpr size_t O_D2     = O_D1     + (size_t)NA * NR * 4;
constexpr size_t O_ALPHA  = O_D2     + (size_t)NA * NR * 4;
constexpr size_t O_BETA   = O_ALPHA  + (size_t)NA * NR * 4;
constexpr size_t O_CLS    = O_BETA   + (size_t)NA * NR * 4;   // (9,4096,21)
constexpr size_t O_ASMC   = O_CLS    + (size_t)NA * NR * NCLS;
constexpr size_t O_BSMC   = O_ASMC   + (size_t)NA * NR;
constexpr size_t O_ALPHAC = O_BSMC   + (size_t)NA * NR;
constexpr size_t O_BETAC  = O_ALPHAC + (size_t)NA * NR;

// ---------------------------------------------------------------------------
// Weight packing kernels (fp32 -> bf16 panels)
// ---------------------------------------------------------------------------
__global__ __launch_bounds__(256) void pack_att_kernel(
    const float* __restrict__ att, bf16* __restrict__ out)
{
    int idx = blockIdx.x * 256 + threadIdx.x;
    if (idx >= DORI * DAPP) return;
    out[idx] = (bf16)att[idx];
}

__global__ __launch_bounds__(256) void pack_smallpad_kernel(
    const float* __restrict__ alpha_w, const float* __restrict__ bbox,
    const float* __restrict__ clsw, const float* __restrict__ alphac,
    bf16* __restrict__ out)                         // [9][2048][128]
{
    int idx = blockIdx.x * 256 + threadIdx.x;
    if (idx >= NA * DORI * SPAD) return;
    int j = idx & (SPAD - 1);
    int k = (idx >> 7) & (DORI - 1);
    int a = idx >> 18;                              // 2048*128 == 1<<18
    float v = 0.0f;
    if (j < 4)        v = alpha_w[((size_t)a * DORI + k) * 4 + j];
    else if (j < 8)   v = bbox[((size_t)a * DORI + k) * 4 + (j - 4)];
    else if (j < 29)  v = clsw[((size_t)a * DORI + k) * NCLS + (j - 8)];
    else if (j == 29) v = alphac[(size_t)a * DORI + k];
    out[idx] = (bf16)v;
}

__global__ __launch_bounds__(256) void pack_qw_kernel(
    const float* __restrict__ w_q, const float* __restrict__ w_q_cls,
    bf16* __restrict__ out)
{
    int idx = blockIdx.x * 256 + threadIdx.x;
    if (idx >= NA * DAPP * QCOLS) return;
    int j = idx % QCOLS;
    int k = (idx / QCOLS) % DAPP;
    int a = idx / (QCOLS * DAPP);
    float v = (j < 128) ? w_q[((size_t)a * DAPP + k) * 128 + j]
                        : w_q_cls[((size_t)a * DAPP + k) * 32 + (j - 128)];
    out[idx] = (bf16)v;
}

__global__ __launch_bounds__(256) void pack_kw_kernel(
    const float* __restrict__ w_k, const float* __restrict__ w_k_cls,
    bf16* __restrict__ out)
{
    int idx = blockIdx.x * 256 + threadIdx.x;
    if (idx >= DAPP * QCOLS) return;
    int j = idx % QCOLS;
    int k = idx / QCOLS;
    float v = (j < 128) ? w_k[(size_t)k * 128 + j]
                        : w_k_cls[(size_t)k * 32 + (j - 128)];
    out[idx] = (bf16)v;
}

// ---------------------------------------------------------------------------
// bf16 WMMA GEMM:  D[z] = op(A[z] @ B[z]);  B panels are pre-packed bf16.
//   BM = 128 (8 waves x 16-row strips), BN in {32,128}, K-step 32.
//   Software-pipelined: double-buffered LDS, global loads of tile kt+1
//   issued before the WMMAs of tile kt, one barrier per K-step.
//   SPLIT: last N-tile of the grid uses the per-anchor small panel B2
//   (ldb2 cols) and routes panel cols [0,32) to fp32 D2 (no relu).
// ---------------------------------------------------------------------------
constexpr int LSTR = 40;   // LDS row stride in bf16 elems (80 B, 16B-aligned rows)

template<int BM, int BN, bool A_BF16, bool RELU, bool OUT_BF16, bool SPLIT>
__global__ __launch_bounds__(256) void wmma_gemm_kernel(
    const void* __restrict__ A, unsigned long long aStride, int lda,
    const bf16* __restrict__ B, unsigned long long bStride, int ldb,
    const bf16* __restrict__ B2, unsigned long long b2Stride, int ldb2,
    void* __restrict__ D, unsigned long long dStride, int ldc,
    float* __restrict__ D2, unsigned long long d2Stride, int ldc2,
    int K)
{
    __shared__ __align__(16) bf16 As[2][BM * LSTR];
    __shared__ __align__(16) bf16 Bt[2][BN * LSTR];

    const int tid  = threadIdx.x;
    const int wave = tid >> 5;
    const int lane = tid & 31;
    const int hlf  = lane >> 4;
    const int r    = lane & 15;

    const int row0 = blockIdx.y * BM;
    const bool small = SPLIT && (blockIdx.x == gridDim.x - 1);

    const char* Ab  = (const char*)A + (size_t)blockIdx.z * aStride * (A_BF16 ? 2 : 4);
    const bf16* BbE = small ? (B2 + (size_t)blockIdx.z * b2Stride)
                            : (B  + (size_t)blockIdx.z * bStride);
    const int   ldbE  = small ? ldb2 : ldb;
    const int   bcol0 = small ? 0 : blockIdx.x * BN;
    char*  Db  = (char*)D + (size_t)blockIdx.z * dStride * (OUT_BF16 ? 2 : 4);
    float* D2b = SPLIT ? (D2 + (size_t)blockIdx.z * d2Stride) : nullptr;

    // register staging
    constexpr int AITER_F = (BM * 8) / 256;            // float4 slots, fp32 A
    constexpr int AITER_H = (BM * 4) / 256;            // uint4 slots, bf16 A
    constexpr int BSLOT_H = BN * 4;                    // uint4 slots, bf16 B
    constexpr int BITER_H = (BSLOT_H + 255) / 256;

    float4 aF[AITER_F];
    uint4  aH[AITER_H];
    uint4  bH[BITER_H];

    // Phase 1: issue all global loads for K-tile at k0 (no LDS traffic).
    auto tile_load = [&](int k0) {
        if (!A_BF16) {
            #pragma unroll
            for (int i = 0; i < AITER_F; ++i) {
                int s = tid + i * 256;
                int row = s >> 3, kc = s & 7;
                aF[i] = *(const float4*)(Ab + ((size_t)(row0 + row) * lda + k0 + kc * 4) * 4);
            }
        } else {
            #pragma unroll
            for (int i = 0; i < AITER_H; ++i) {
                int s = tid + i * 256;
                int row = s >> 2, kc = s & 3;
                aH[i] = *(const uint4*)(Ab + ((size_t)(row0 + row) * lda + k0 + kc * 8) * 2);
            }
        }
        #pragma unroll
        for (int i = 0; i < BITER_H; ++i) {
            int s = tid + i * 256;
            if (s < BSLOT_H) {
                int n = (s % (BN / 8)) * 8, k = s / (BN / 8);
                bH[i] = *(const uint4*)((const char*)BbE
                            + ((size_t)(k0 + k) * ldbE + bcol0 + n) * 2);
            }
        }
    };

    // Phase 2: convert and store staged tile into LDS buffers.
    auto tile_store = [&](bf16* __restrict__ Ad, bf16* __restrict__ Bd) {
        if (!A_BF16) {
            #pragma unroll
            for (int i = 0; i < AITER_F; ++i) {
                int s = tid + i * 256;
                int row = s >> 3, kc = s & 7;
                H4 p;
                p.h[0] = (bf16)aF[i].x; p.h[1] = (bf16)aF[i].y;
                p.h[2] = (bf16)aF[i].z; p.h[3] = (bf16)aF[i].w;
                *(unsigned long long*)&Ad[row * LSTR + kc * 4] = p.q;   // ds_store_b64
            }
        } else {
            #pragma unroll
            for (int i = 0; i < AITER_H; ++i) {
                int s = tid + i * 256;
                int row = s >> 2, kc = s & 3;
                *(uint4*)&Ad[row * LSTR + kc * 8] = aH[i];              // ds_store_b128
            }
        }
        #pragma unroll
        for (int i = 0; i < BITER_H; ++i) {
            int s = tid + i * 256;
            if (s < BSLOT_H) {
                int n = (s % (BN / 8)) * 8, k = s / (BN / 8);
                const bf16* hp = (const bf16*)&bH[i];
                #pragma unroll
                for (int j = 0; j < 8; ++j) Bd[(n + j) * LSTR + k] = hp[j];
            }
        }
    };

    v8f acc[BN / 16] = {};
    const int KT = K >> 5;

    tile_load(0);
    tile_store(As[0], Bt[0]);
    __syncthreads();

    for (int kt = 0; kt < KT; ++kt) {
        const int cur = kt & 1;

        if (kt + 1 < KT) {
            if (kt + 2 < KT) {   // prefetch 2 tiles ahead into near caches
                const char* pf = Ab + ((size_t)(row0 + (tid & (BM - 1))) * lda
                                       + ((kt + 2) << 5)) * (A_BF16 ? 2 : 4);
                __builtin_prefetch(pf, 0, 3);
            }
            tile_load((kt + 1) << 5);      // in flight during the WMMAs below
        }

        // ---- fragments + WMMA on the current LDS tile ----
        AFrag fa;
        const bf16* pa = &As[cur][(wave * 16 + r) * LSTR + hlf * 8];
        fa.u[0] = *(const uint4*)pa;
        fa.u[1] = *(const uint4*)(pa + 16);

        AFrag fb[BN / 16];
        #pragma unroll
        for (int nt = 0; nt < BN / 16; ++nt) {
            const bf16* pb = &Bt[cur][(nt * 16 + r) * LSTR + hlf * 8];
            fb[nt].u[0] = *(const uint4*)pb;
            fb[nt].u[1] = *(const uint4*)(pb + 16);
        }
        #pragma unroll
        for (int nt = 0; nt < BN / 16; ++nt) {
            acc[nt] = __builtin_amdgcn_wmma_f32_16x16x32_bf16(
                false, fa.v, false, fb[nt].v, (short)0, acc[nt], false, false);
        }

        if (kt + 1 < KT) tile_store(As[cur ^ 1], Bt[cur ^ 1]);
        __syncthreads();
    }

    // ---- epilogue: C/D layout -> (m = v + 8*hlf, n = r) per 16x16 tile ----
    #pragma unroll
    for (int nt = 0; nt < BN / 16; ++nt) {
        #pragma unroll
        for (int v = 0; v < 8; ++v) {
            float x = acc[nt][v];
            size_t row = (size_t)row0 + wave * 16 + v + 8 * hlf;
            int coll = nt * 16 + r;
            if (!small) {
                if (RELU) x = fmaxf(x, 0.0f);
                size_t col = (size_t)blockIdx.x * BN + coll;
                if (OUT_BF16) *(bf16*)(Db + (row * ldc + col) * 2) = (bf16)x;
                else          *(float*)(Db + (row * ldc + col) * 4) = x;
            } else {
                if (coll < SCOLS) D2b[row * ldc2 + coll] = x;   // no relu, fp32
            }
        }
    }
}

// ---------------------------------------------------------------------------
// Epilogue: beta dots, softmaxes over the 9-anchor axis, weighted sums,
// and scatter of all 13 concatenated outputs. One thread per roi.
// ---------------------------------------------------------------------------
__global__ __launch_bounds__(256) void epilogue_kernel(
    const float* __restrict__ S,          // [9][4096][32]
    const float* __restrict__ Q,          // [9][4096][160]
    const float* __restrict__ KY,         // [4096][160]
    const float* __restrict__ ious,       // [9][1][4096]
    const float* __restrict__ delta_rois, // [9][4096][5]
    float* __restrict__ out)
{
    int n = blockIdx.x * 256 + threadIdx.x;
    if (n >= NR) return;
    const float scale = 0.17677669529663687f;   // 1/sqrt(32)

    float alpha[NA][4], beta[NA][4], dpo[NA][4], alphac[NA], betac[NA];
    const float* ky = KY + (size_t)n * QCOLS;

    #pragma unroll
    for (int a = 0; a < NA; ++a) {
        const float* q = Q + ((size_t)a * NR + n) * QCOLS;
        float d[4] = {0.f, 0.f, 0.f, 0.f}, dc = 0.f;
        #pragma unroll
        for (int kk = 0; kk < 32; ++kk) {
            #pragma unroll
            for (int c = 0; c < 4; ++c) d[c] += q[kk * 4 + c] * ky[kk * 4 + c];
            dc += q[128 + kk] * ky[128 + kk];
        }
        float iou = ious[(size_t)a * NR + n];
        const float* s  = S + ((size_t)a * NR + n) * SCOLS;
        const float* dr = delta_rois + ((size_t)a * NR + n) * 5;
        #pragma unroll
        for (int c = 0; c < 4; ++c) {
            beta[a][c]  = d[c] * scale - iou;
            alpha[a][c] = s[c];
            dpo[a][c]   = dr[1 + c] + s[4 + c];
        }
        betac[a]  = dc * scale - iou;
        alphac[a] = s[29];

        size_t b4 = ((size_t)a * NR + n) * 4;
        #pragma unroll
        for (int c = 0; c < 4; ++c) {
            out[O_ALPHA + b4 + c] = alpha[a][c];
            out[O_BETA  + b4 + c] = beta[a][c];
            out[O_D1    + b4 + c] = dpo[a][c];
            out[O_D2    + b4 + c] = dpo[a][c];
        }
        size_t b21 = ((size_t)a * NR + n) * (size_t)NCLS;
        #pragma unroll
        for (int j = 0; j < NCLS; ++j) out[O_CLS + b21 + j] = s[8 + j];
        out[O_ALPHAC + (size_t)a * NR + n] = alphac[a];
        out[O_BETAC  + (size_t)a * NR + n] = betac[a];
    }

    float outv[4] = {0.f, 0.f, 0.f, 0.f}, outb[4] = {0.f, 0.f, 0.f, 0.f};
    #pragma unroll
    for (int c = 0; c < 4; ++c) {
        float mA = -3.4e38f, mB = -3.4e38f;
        #pragma unroll
        for (int a = 0; a < NA; ++a) { mA = fmaxf(mA, alpha[a][c]); mB = fmaxf(mB, beta[a][c]); }
        float eA[NA], eB[NA], sA = 0.f, sB = 0.f;
        #pragma unroll
        for (int a = 0; a < NA; ++a) {
            eA[a] = __expf(alpha[a][c] - mA); sA += eA[a];
            eB[a] = __expf(beta[a][c]  - mB); sB += eB[a];
        }
        float rA = 1.0f / sA, rB = 1.0f / sB;
        #pragma unroll
        for (int a = 0; a < NA; ++a) {
            float va = eA[a] * rA, vb = eB[a] * rB;
            size_t b4 = ((size_t)a * NR + n) * 4 + c;
            out[O_ASM + b4] = va;
            out[O_BSM + b4] = vb;
            outv[c] += dpo[a][c] * va;
            outb[c] += dpo[a][c] * vb;
        }
    }
    #pragma unroll
    for (int c = 0; c < 4; ++c) {
        out[O_OUT  + (size_t)n * 4 + c] = outv[c];
        out[O_OUTB + (size_t)n * 4 + c] = outb[c];
    }

    float mAc = -3.4e38f, mBc = -3.4e38f;
    #pragma unroll
    for (int a = 0; a < NA; ++a) { mAc = fmaxf(mAc, alphac[a]); mBc = fmaxf(mBc, betac[a]); }
    float eAc[NA], eBc[NA], sAc = 0.f, sBc = 0.f;
    #pragma unroll
    for (int a = 0; a < NA; ++a) {
        eAc[a] = __expf(alphac[a] - mAc); sAc += eAc[a];
        eBc[a] = __expf(betac[a]  - mBc); sBc += eBc[a];
    }
    float rAc = 1.0f / sAc, rBc = 1.0f / sBc;
    #pragma unroll
    for (int a = 0; a < NA; ++a) {
        out[O_ASMC + (size_t)a * NR + n] = eAc[a] * rAc;
        out[O_BSMC + (size_t)a * NR + n] = eBc[a] * rBc;
    }
}

// ---------------------------------------------------------------------------
extern "C" void kernel_launch(void* const* d_in, const int* in_sizes, int n_in,
                              void* d_out, int out_size, void* d_ws, size_t ws_size,
                              hipStream_t stream) {
    const float* delta_rois     = (const float*)d_in[1];
    const float* features       = (const float*)d_in[2];
    const float* gt_features    = (const float*)d_in[3];
    const float* ious           = (const float*)d_in[4];
    const float* attention_feat = (const float*)d_in[5];
    const float* w_k            = (const float*)d_in[6];
    const float* w_q            = (const float*)d_in[7];
    const float* alpha_w        = (const float*)d_in[8];
    const float* bbox_regress   = (const float*)d_in[9];
    const float* w_k_cls        = (const float*)d_in[10];
    const float* w_q_cls        = (const float*)d_in[11];
    const float* alpha_w_cls    = (const float*)d_in[12];
    const float* cls_score_w    = (const float*)d_in[13];

    char* ws = (char*)d_ws;
    bf16*  off  = (bf16*)(ws + WS_OFF);
    float* S    = (float*)(ws + WS_S);
    float* Q    = (float*)(ws + WS_Q);
    float* KY   = (float*)(ws + WS_K);
    bf16*  Batt = (bf16*)(ws + WS_BATT);
    bf16*  Wsmp = (bf16*)(ws + WS_WSMP);
    bf16*  Wq   = (bf16*)(ws + WS_WQ);
    bf16*  Wk   = (bf16*)(ws + WS_WK);
    float* out  = (float*)d_out;

    // weight packing (tiny, L2-resident)
    pack_att_kernel<<<(DORI * DAPP + 255) / 256, 256, 0, stream>>>(attention_feat, Batt);
    pack_smallpad_kernel<<<(NA * DORI * SPAD + 255) / 256, 256, 0, stream>>>(
        alpha_w, bbox_regress, cls_score_w, alpha_w_cls, Wsmp);
    pack_qw_kernel<<<(NA * DAPP * QCOLS + 255) / 256, 256, 0, stream>>>(w_q, w_q_cls, Wq);
    pack_kw_kernel<<<(DAPP * QCOLS + 255) / 256, 256, 0, stream>>>(w_k, w_k_cls, Wk);

    // Stage 1+2 fused: one pass over features per anchor.
    //   N-tiles 0..3 -> off = relu(features @ att) (bf16)
    //   N-tile  4    -> S   = features @ Wsmp[:, :32] (fp32)
    wmma_gemm_kernel<128, 128, false, true, true, true>
        <<<dim3(DAPP / 128 + 1, NR / 128, NA), 256, 0, stream>>>(
            features, (size_t)NR * DORI, DORI,
            Batt, 0, DAPP,
            Wsmp, (size_t)DORI * SPAD, SPAD,
            off, (size_t)NR * DAPP, DAPP,
            S, (size_t)NR * SCOLS, SCOLS,
            DORI);
    // gt: off[9] = relu(gt_features @ att)
    wmma_gemm_kernel<128, 128, false, true, true, false>
        <<<dim3(DAPP / 128, NR / 128, 1), 256, 0, stream>>>(
            gt_features, 0, DORI,
            Batt, 0, DAPP,
            nullptr, 0, 0,
            off + (size_t)NA * NR * DAPP, 0, DAPP,
            nullptr, 0, 0,
            DORI);

    // Stage 3: Q = off @ Wq (per anchor), KY = gt_off @ Wk
    wmma_gemm_kernel<128, 32, true, false, false, false>
        <<<dim3(QCOLS / 32, NR / 128, NA), 256, 0, stream>>>(
            off, (size_t)NR * DAPP, DAPP,
            Wq, (size_t)DAPP * QCOLS, QCOLS,
            nullptr, 0, 0,
            Q, (size_t)NR * QCOLS, QCOLS,
            nullptr, 0, 0,
            DAPP);
    wmma_gemm_kernel<128, 32, true, false, false, false>
        <<<dim3(QCOLS / 32, NR / 128, 1), 256, 0, stream>>>(
            off + (size_t)NA * NR * DAPP, 0, DAPP,
            Wk, 0, QCOLS,
            nullptr, 0, 0,
            KY, 0, QCOLS,
            nullptr, 0, 0,
            DAPP);

    // Stage 4: dots, softmaxes, outputs
    epilogue_kernel<<<NR / 256, 256, 0, stream>>>(S, Q, KY, ious, delta_rois, out);
}
